// LSTMCell_32933809225810
// MI455X (gfx1250) — compile-verified, matrix operand order checked by reading, not run on hardware
//
#include <hip/hip_runtime.h>
#include <hip/hip_bf16.h>

// ---------------------------------------------------------------------------
// LSTM cell for MI455X (gfx1250, wave32, WMMA + TDM).
//   gates = [x|h] @ [W_I|W_F|W_G|W_O] + b  -> fused epilogue -> H, C
// Pass 1: cvt_x : fp32 [x|h]   -> bf16 Xb[8192][2048]          (d_ws)
// Pass 2: cvt_w : fp32 W[k][n] -> bf16 WT[4][1024 n][2048 k]   (d_ws)
// Pass 3: GEMM  : bf16 WMMA 16x16x32, fp32 accum, fused LSTM epilogue.
//         Tiles staged to LDS by the Tensor Data Mover (tensor_load_to_lds,
//         6-arg clang-23 form), double-buffered, one barrier per K-iteration.
// Block tile: BM=128 x BN=32 x 4 gates, BK=64; 256 threads = 8 waves (4Mx2N).
// ---------------------------------------------------------------------------

typedef __attribute__((ext_vector_type(16))) __bf16 v16bf;
typedef __attribute__((ext_vector_type(8)))  __bf16 v8bf;
typedef __attribute__((ext_vector_type(8)))  float  v8f;

#define DIM   1024
#define KDIM  2048
#define BATCH 8192
#define BM    128
#define BN    32
#define BK    64

#if __has_builtin(__builtin_amdgcn_tensor_load_to_lds) && \
    __has_builtin(__builtin_amdgcn_s_wait_tensorcnt)
#define USE_TDM 1
#else
#define USE_TDM 0
#endif

__device__ __forceinline__ unsigned pk2bf(float a, float b) {
    unsigned short ul = __builtin_bit_cast(unsigned short, (__bf16)a);
    unsigned short uh = __builtin_bit_cast(unsigned short, (__bf16)b);
    return (unsigned)ul | ((unsigned)uh << 16);
}

__device__ __forceinline__ float sigmoidf_(float x) {
    return 1.0f / (1.0f + __expf(-x));
}

// ---- pass 1: [x|h] fp32 -> Xb bf16, row-major [BATCH][KDIM] ---------------
__global__ __launch_bounds__(256)
void cvt_x_kernel(const float* __restrict__ x, const float* __restrict__ h,
                  unsigned short* __restrict__ Xb)
{
    const size_t idx = (size_t)blockIdx.x * 256 + threadIdx.x; // 8 elems each
    const int r  = (int)(idx >> 8);
    const int k8 = (int)(idx & 255) << 3;
    const float* src = (k8 < DIM) ? (x + (size_t)r * DIM + k8)
                                  : (h + (size_t)r * DIM + (k8 - DIM));
    const float4 a = ((const float4*)src)[0];
    const float4 b = ((const float4*)src)[1];
    uint4 o;
    o.x = pk2bf(a.x, a.y);
    o.y = pk2bf(a.z, a.w);
    o.z = pk2bf(b.x, b.y);
    o.w = pk2bf(b.z, b.w);
    *(uint4*)(Xb + (size_t)r * KDIM + k8) = o;
}

// ---- pass 2: W[k][n] fp32 -> WT[g][n][k] bf16 (transpose via LDS) ---------
__global__ __launch_bounds__(256)
void cvt_w_kernel(const float* __restrict__ WI, const float* __restrict__ WF,
                  const float* __restrict__ WG, const float* __restrict__ WO,
                  unsigned short* __restrict__ WT)
{
    __shared__ unsigned short tile[32][33];
    const int kBlk = blockIdx.x * 32;
    const int nBlk = blockIdx.y * 32;
    const int g    = blockIdx.z;
    const float* W = (g == 0) ? WI : (g == 1) ? WF : (g == 2) ? WG : WO;

    const int tid = threadIdx.x;
    {
        const int k  = tid >> 3;
        const int n4 = (tid & 7) << 2;
        const float4 v = *(const float4*)(W + (size_t)(kBlk + k) * DIM + nBlk + n4);
        tile[n4 + 0][k] = (unsigned short)(pk2bf(v.x, v.x) & 0xFFFFu);
        tile[n4 + 1][k] = (unsigned short)(pk2bf(v.y, v.y) & 0xFFFFu);
        tile[n4 + 2][k] = (unsigned short)(pk2bf(v.z, v.z) & 0xFFFFu);
        tile[n4 + 3][k] = (unsigned short)(pk2bf(v.w, v.w) & 0xFFFFu);
    }
    __syncthreads();
    {
        const int n  = tid >> 3;
        const int k4 = (tid & 7) << 2;
        unsigned lo = (unsigned)tile[n][k4 + 0] | ((unsigned)tile[n][k4 + 1] << 16);
        unsigned hi = (unsigned)tile[n][k4 + 2] | ((unsigned)tile[n][k4 + 3] << 16);
        *(uint2*)(WT + (((size_t)g * DIM + nBlk + n) * KDIM) + kBlk + k4)
            = make_uint2(lo, hi);
    }
}

// ---- TDM descriptor build + issue (CDNA5 D#, ISA 08_async_tensor §8) ------
#if USE_TDM
typedef __attribute__((ext_vector_type(4))) unsigned int u32x4;
typedef __attribute__((ext_vector_type(8))) int          i32x8;
typedef __attribute__((ext_vector_type(4))) int          i32x4;

__device__ __forceinline__ unsigned lds_byte_off(const void* p) {
    // generic LDS address: low 32 bits are the LDS byte offset
    return (unsigned)(unsigned long long)(size_t)p;
}

__device__ __forceinline__ void tdm_load(unsigned lds_addr, const void* gptr,
                                         unsigned tile_d0, unsigned tile_d1,
                                         unsigned tile_d2,
                                         unsigned long long tensor_d0,
                                         unsigned long long tensor_d1,
                                         unsigned tensor_d2,
                                         unsigned long long stride0,
                                         unsigned long long stride1)
{
    const unsigned long long ga = (unsigned long long)(size_t)gptr;
    u32x4 g0;
    g0.x = 1u;                                                // count=1, load
    g0.y = lds_addr;                                          // D#.lds_addr
    g0.z = (unsigned)(ga & 0xFFFFFFFFu);                      // global_addr lo
    g0.w = (unsigned)((ga >> 32) & 0x01FFFFFFu) | (2u << 30); // addr hi | type=2
    i32x8 g1;
    g1[0] = (int)(1u << 16);                                  // data_size=1 (2B)
    g1[1] = (int)((tensor_d0 & 0xFFFFu) << 16);               // tensor_dim0[15:0]
    g1[2] = (int)(((tensor_d0 >> 16) & 0xFFFFu) |
                  ((tensor_d1 & 0xFFFFu) << 16));             // d0 hi | d1 lo
    g1[3] = (int)(((tensor_d1 >> 16) & 0xFFFFu) |
                  (tile_d0 << 16));                           // d1 hi | tile_dim0
    g1[4] = (int)(tile_d1 | (tile_d2 << 16));                 // tile_dim1|tile_dim2
    g1[5] = (int)(stride0 & 0xFFFFFFFFu);                     // dim0_stride lo
    g1[6] = (int)(((stride0 >> 32) & 0xFFFFu) |
                  ((stride1 & 0xFFFFu) << 16));               // s0 hi | s1 lo
    g1[7] = (int)((stride1 >> 16) & 0xFFFFFFFFu);             // s1 hi
    i32x4 g2;
    g2[0] = (int)tensor_d2;                                   // tensor_dim2
    g2[1] = 0; g2[2] = 0; g2[3] = 0;
    i32x4 g3;
    g3[0] = 0; g3[1] = 0; g3[2] = 0; g3[3] = 0;
    i32x8 g4;                                                 // zero-filled (6-arg form)
    g4[0] = 0; g4[1] = 0; g4[2] = 0; g4[3] = 0;
    g4[4] = 0; g4[5] = 0; g4[6] = 0; g4[7] = 0;
    __builtin_amdgcn_tensor_load_to_lds(g0, g1, g2, g3, g4, 0);
}
#endif

// ---- 16 WMMAs on one staged tile pair -------------------------------------
__device__ __forceinline__ void compute_tile(const unsigned short* sXb,
                                             const unsigned short* sWb,
                                             int wm, int wn, int lrow, int koff,
                                             v8f acc[4][2])
{
    #pragma unroll
    for (int kk = 0; kk < BK; kk += 32) {
        v16bf aF[2];
        #pragma unroll
        for (int t = 0; t < 2; ++t) {
            const int row = wm * 32 + t * 16 + lrow;
            const v8bf lo = *(const v8bf*)((const __bf16*)sXb + row * BK + kk + koff);
            const v8bf hi = *(const v8bf*)((const __bf16*)sXb + row * BK + kk + koff + 16);
            aF[t] = __builtin_shufflevector(lo, hi,
                        0,1,2,3,4,5,6,7,8,9,10,11,12,13,14,15);
        }
        #pragma unroll
        for (int g = 0; g < 4; ++g) {
            const int n = wn * 16 + lrow;
            const v8bf lo = *(const v8bf*)((const __bf16*)sWb + (g * BN + n) * BK + kk + koff);
            const v8bf hi = *(const v8bf*)((const __bf16*)sWb + (g * BN + n) * BK + kk + koff + 16);
            const v16bf bFrag = __builtin_shufflevector(lo, hi,
                        0,1,2,3,4,5,6,7,8,9,10,11,12,13,14,15);
            acc[g][0] = __builtin_amdgcn_wmma_f32_16x16x32_bf16(
                            false, aF[0], false, bFrag, (short)0, acc[g][0], false, false);
            acc[g][1] = __builtin_amdgcn_wmma_f32_16x16x32_bf16(
                            false, aF[1], false, bFrag, (short)0, acc[g][1], false, false);
        }
    }
}

// ---- pass 3: bf16 WMMA GEMM + fused LSTM epilogue -------------------------
__global__ __launch_bounds__(256)
void lstm_gemm_kernel(const unsigned short* __restrict__ Xb,
                      const unsigned short* __restrict__ WT,
                      const float* __restrict__ c,
                      const float* __restrict__ bI, const float* __restrict__ bF,
                      const float* __restrict__ bG, const float* __restrict__ bO,
                      float* __restrict__ Hout, float* __restrict__ Cout)
{
    const int tid  = threadIdx.x;
    const int lane = tid & 31;
    const int wv   = tid >> 5;
    const int wm   = wv & 3;
    const int wn   = wv >> 2;

    const int nBlk = blockIdx.x * BN;
    const int mBlk = blockIdx.y * BM;

    v8f acc[4][2];
    #pragma unroll
    for (int g = 0; g < 4; ++g) { acc[g][0] = {}; acc[g][1] = {}; }

    const int rsel = lane >> 4;
    const int koff = rsel ? 8 : 0;
    const int lrow = lane & 15;

#if USE_TDM
    // double-buffered LDS, filled by the Tensor Data Mover
    __shared__ unsigned short sX[2][BM * BK];       // 2 x 16 KB
    __shared__ unsigned short sW[2][4 * BN * BK];   // 2 x 16 KB

    const unsigned short* Xrow = Xb + (size_t)mBlk * KDIM;  // [128][2048]
    const unsigned short* Wrow = WT + (size_t)nBlk * KDIM;  // [n][2048], gates 1024 rows apart

    if (wv == 0) {
        // prologue: fill buffer 0 (X: 2D tile 64k x 128r; W: 3D tile 64k x 32n x 4g)
        tdm_load(lds_byte_off(&sX[0][0]), Xrow,
                 BK, BM, 0, KDIM, BATCH, 0, KDIM, 0);
        tdm_load(lds_byte_off(&sW[0][0]), Wrow,
                 BK, BN, 4, KDIM, DIM, 4, KDIM, (unsigned long long)DIM * KDIM);
    }

    const int NIT = KDIM / BK;   // 32
    for (int it = 0; it < NIT; ++it) {
        const int cur = it & 1;
        const int nxt = cur ^ 1;
        if (wv == 0) {
            __builtin_amdgcn_s_wait_tensorcnt(0);   // current buffer DMA done
        }
        __syncthreads();                            // publish LDS to all waves
        if (wv == 0 && (it + 1) < NIT) {
            const size_t kb = (size_t)(it + 1) * BK;
            tdm_load(lds_byte_off(&sX[nxt][0]), Xrow + kb,
                     BK, BM, 0, KDIM, BATCH, 0, KDIM, 0);
            tdm_load(lds_byte_off(&sW[nxt][0]), Wrow + kb,
                     BK, BN, 4, KDIM, DIM, 4, KDIM, (unsigned long long)DIM * KDIM);
        }
        compute_tile(&sX[cur][0], &sW[cur][0], wm, wn, lrow, koff, acc);
        // WAR on buf[cur] vs TDM at it+1 is ordered by the barrier at it+1
    }
#else
    // fallback: manual staging, single buffer, two barriers per iteration
    __shared__ unsigned short sX[1][BM * BK];
    __shared__ unsigned short sW[1][4 * BN * BK];

    for (int k0 = 0; k0 < KDIM; k0 += BK) {
        #pragma unroll
        for (int it = 0; it < 4; ++it) {
            const int idx = tid + it * 256;
            const int r   = idx >> 3;
            const int k8  = (idx & 7) << 3;
            const uint4 v = *(const uint4*)(Xb + (size_t)(mBlk + r) * KDIM + k0 + k8);
            *(uint4*)(&sX[0][r * BK + k8]) = v;
        }
        #pragma unroll
        for (int it = 0; it < 4; ++it) {
            const int idx = tid + it * 256;
            const int g   = idx >> 8;
            const int n   = (idx >> 3) & 31;
            const int k8  = (idx & 7) << 3;
            const uint4 v = *(const uint4*)(
                WT + ((size_t)g * DIM + nBlk + n) * KDIM + k0 + k8);
            *(uint4*)(&sW[0][(g * BN + n) * BK + k8]) = v;
        }
        __syncthreads();
        compute_tile(&sX[0][0], &sW[0][0], wm, wn, lrow, koff, acc);
        __syncthreads();
    }
#endif

    // ---- fused LSTM epilogue ----------------------------------------------
    const int col = nBlk + wn * 16 + lrow;
    const float biasI = bI[col];
    const float biasF = bF[col];
    const float biasG = bG[col];
    const float biasO = bO[col];

    #pragma unroll
    for (int t = 0; t < 2; ++t) {
        #pragma unroll
        for (int j = 0; j < 8; ++j) {
            const int row = mBlk + wm * 32 + t * 16 + (rsel << 3) + j;
            const size_t off = (size_t)row * DIM + col;
            const float iv = sigmoidf_(acc[0][t][j] + biasI);
            const float fv = sigmoidf_(acc[1][t][j] + biasF);
            const float gv = tanhf(acc[2][t][j] + biasG);
            const float ov = sigmoidf_(acc[3][t][j] + biasO);
            const float cn = fv * c[off] + iv * gv;
            Hout[off] = ov * tanhf(cn);
            Cout[off] = cn;
        }
    }
}

extern "C" void kernel_launch(void* const* d_in, const int* in_sizes, int n_in,
                              void* d_out, int out_size, void* d_ws, size_t ws_size,
                              hipStream_t stream) {
    const float* x  = (const float*)d_in[0];
    const float* h  = (const float*)d_in[1];
    const float* c  = (const float*)d_in[2];
    const float* WI = (const float*)d_in[3];
    const float* bI = (const float*)d_in[4];
    const float* WF = (const float*)d_in[5];
    const float* bF = (const float*)d_in[6];
    const float* WG = (const float*)d_in[7];
    const float* bG = (const float*)d_in[8];
    const float* WO = (const float*)d_in[9];
    const float* bO = (const float*)d_in[10];

    // workspace: Xb bf16 [8192][2048] (32 MB) + WT bf16 [4][1024][2048] (16 MB)
    unsigned short* Xb = (unsigned short*)d_ws;
    unsigned short* WT = Xb + (size_t)BATCH * KDIM;

    float* Hout = (float*)d_out;
    float* Cout = Hout + (size_t)BATCH * DIM;

    cvt_x_kernel<<<(BATCH * (KDIM / 8)) / 256, 256, 0, stream>>>(x, h, Xb);
    cvt_w_kernel<<<dim3(KDIM / 32, DIM / 32, 4), 256, 0, stream>>>(WI, WF, WG, WO, WT);

    dim3 grid(DIM / BN, BATCH / BM);   // (32, 64) = 2048 blocks
    lstm_gemm_kernel<<<grid, 256, 0, stream>>>(Xb, WT, c, bI, bF, bG, bO, Hout, Cout);
}
